// WARPLoss_89146341196222
// MI455X (gfx1250) — matrix operand order repaired and checked by reading, not compiled, not run on hardware
//
#include <hip/hip_runtime.h>
#include <hip/hip_bf16.h>
#include <cstdint>

#define N_LBL   89
#define BATCH   1024
#define WEMB    300
#define IN_DIM  1324
#define D_PAD   1344   // 42 chunks of 32
#define E_PAD   320    // 20 tiles of 16
#define MEN_PD  320
#define N_CHUNK 42

typedef float    f32x4  __attribute__((ext_vector_type(4)));
typedef unsigned u32x4  __attribute__((ext_vector_type(4)));
typedef float    v8f    __attribute__((ext_vector_type(8)));
typedef __bf16   v16bf  __attribute__((ext_vector_type(16)));

// ---- split f32 into bf16 hi + bf16 lo (RNE both) ----
__device__ __forceinline__ void split_bf16(float x, unsigned short& hi, unsigned short& lo) {
  unsigned u = __float_as_uint(x);
  unsigned r = u + 0x7FFFu + ((u >> 16) & 1u);
  hi = (unsigned short)(r >> 16);
  float hif = __uint_as_float(((unsigned)hi) << 16);
  float lof = x - hif;
  unsigned ul = __float_as_uint(lof);
  unsigned rl = ul + 0x7FFFu + ((ul >> 16) & 1u);
  lo = (unsigned short)(rl >> 16);
}

// =================== Kernel 0: pad + split precision prep ===================
__global__ __launch_bounds__(256) void prep_kernel(
    const float* __restrict__ W, const float* __restrict__ men,
    const float* __restrict__ bvec, const float* __restrict__ te,
    unsigned short* __restrict__ Whi, unsigned short* __restrict__ Wlo,
    float* __restrict__ men_pad, float* __restrict__ b_pad, float* __restrict__ te_pad)
{
  const int NW = E_PAD * D_PAD;       // 430080
  const int NM = BATCH * MEN_PD;      // 327680
  const int NB = E_PAD;               // 320
  const int NT = N_LBL * E_PAD;       // 28480
  int i = blockIdx.x * blockDim.x + threadIdx.x;
  if (i < NW) {
    int e = i / D_PAD, d = i % D_PAD;
    float v = (e < WEMB && d < IN_DIM) ? W[e * IN_DIM + d] : 0.0f;
    unsigned short h, l; split_bf16(v, h, l);
    Whi[i] = h; Wlo[i] = l;
  } else if (i < NW + NM) {
    int j = i - NW; int b = j / MEN_PD, c = j % MEN_PD;
    men_pad[j] = (c < WEMB) ? men[b * WEMB + c] : 0.0f;
  } else if (i < NW + NM + NB) {
    int e = i - NW - NM;
    b_pad[e] = (e < WEMB) ? bvec[e] : 0.0f;
  } else if (i < NW + NM + NB + NT) {
    int j = i - NW - NM - NB; int t = j / E_PAD, e = j % E_PAD;
    te_pad[j] = (e < WEMB) ? te[t * WEMB + e] : 0.0f;
  }
}

// =================== Kernel 1: fused GEMM + relu + type-dot (WMMA) ===================
// grid = (16 batch-tiles of 64, 89 labels); 256 threads = 8 waves
// wave = (msub 0..3) x (egroup 0..1); each wave: 16 rows x 10 e-tiles, full K loop.
__global__ __launch_bounds__(256) void score_kernel(
    const float* __restrict__ ctx,
    const unsigned short* __restrict__ Whi, const unsigned short* __restrict__ Wlo,
    const float* __restrict__ men_pad, const float* __restrict__ b_pad,
    const float* __restrict__ te_pad, float* __restrict__ scores)
{
  const int t   = blockIdx.y;
  const int b0  = blockIdx.x * 64;
  const int tid = threadIdx.x;
  const int wave = tid >> 5, lane = tid & 31;
  const int msub = wave & 3, eg = wave >> 2;
  const int lhalf = lane >> 4, l15 = lane & 15;
  const int row = b0 + msub * 16 + l15;        // A-fragment source row for this lane
  const int kb  = lhalf * 8;                   // A: K octet select per lane-half
  const int kb2 = lhalf * 8;                   // B: K octet select per lane-half

  v8f acc[10];
  #pragma unroll
  for (int j = 0; j < 10; ++j)
    #pragma unroll
    for (int r = 0; r < 8; ++r) acc[j][r] = 0.0f;

  const float* ctxrow = ctx + ((size_t)t * BATCH + row) * 1024;
  const float* menrow = men_pad + (size_t)row * MEN_PD;

  for (int chunk = 0; chunk < N_CHUNK; ++chunk) {
    // ---- A fragment: 16 f32 -> bf16 hi/lo, ISA 16-bit A layout ----
    const float* ap = (chunk < 32) ? (ctxrow + chunk * 32 + kb)
                                   : (menrow + chunk * 32 + kb - 1024);
    f32x4 a0 = *(const f32x4*)(ap);
    f32x4 a1 = *(const f32x4*)(ap + 4);
    f32x4 a2 = *(const f32x4*)(ap + 16);
    f32x4 a3 = *(const f32x4*)(ap + 20);
    float xs[16];
    #pragma unroll
    for (int q = 0; q < 4; ++q) { xs[q] = a0[q]; xs[4+q] = a1[q]; xs[8+q] = a2[q]; xs[12+q] = a3[q]; }
    union AV { v16bf v; unsigned u[8]; } ahi, alo;
    #pragma unroll
    for (int q = 0; q < 8; ++q) {
      unsigned short h0, l0, h1, l1;
      split_bf16(xs[2*q],   h0, l0);
      split_bf16(xs[2*q+1], h1, l1);
      ahi.u[q] = (unsigned)h0 | ((unsigned)h1 << 16);
      alo.u[q] = (unsigned)l0 | ((unsigned)l1 << 16);
    }
    // ---- B fragments straight from packed bf16 W (zero-padded) ----
    #pragma unroll
    for (int j = 0; j < 10; ++j) {
      const int e = (eg * 10 + j) * 16 + l15;
      const size_t boff = (size_t)e * D_PAD + (size_t)chunk * 32 + kb2;
      union BV { v16bf v; u32x4 q[2]; } bhi, blo;
      bhi.q[0] = *(const u32x4*)(Whi + boff);
      bhi.q[1] = *(const u32x4*)(Whi + boff + 16);
      blo.q[0] = *(const u32x4*)(Wlo + boff);
      blo.q[1] = *(const u32x4*)(Wlo + boff + 16);
      // split-precision f32 emulation: hi*hi + hi*lo + lo*hi
      acc[j] = __builtin_amdgcn_wmma_f32_16x16x32_bf16(false, ahi.v, false, bhi.v,
                                                       (short)0, acc[j], false, false);
      acc[j] = __builtin_amdgcn_wmma_f32_16x16x32_bf16(false, ahi.v, false, blo.v,
                                                       (short)0, acc[j], false, false);
      acc[j] = __builtin_amdgcn_wmma_f32_16x16x32_bf16(false, alo.v, false, bhi.v,
                                                       (short)0, acc[j], false, false);
    }
  }

  // ---- epilogue: relu(acc + bias) * types_emb, reduce over e ----
  float partial[8];
  #pragma unroll
  for (int r = 0; r < 8; ++r) partial[r] = 0.0f;
  #pragma unroll
  for (int j = 0; j < 10; ++j) {
    const int e = (eg * 10 + j) * 16 + l15;   // C/D layout: N = lane&15
    const float bias = b_pad[e];
    const float tev  = te_pad[t * E_PAD + e];
    #pragma unroll
    for (int r = 0; r < 8; ++r) {
      float h = acc[j][r] + bias;
      h = (h > 0.0f) ? h : 0.0f;
      partial[r] += h * tev;
    }
  }

  __shared__ float sums[64];
  if (tid < 64) sums[tid] = 0.0f;
  __syncthreads();
  #pragma unroll
  for (int r = 0; r < 8; ++r) {
    float v = partial[r];
    v += __shfl_xor(v, 1, 32);
    v += __shfl_xor(v, 2, 32);
    v += __shfl_xor(v, 4, 32);
    v += __shfl_xor(v, 8, 32);                 // reduce across 16 N-lanes per half
    if (l15 == 0)                              // C/D layout: M = r + 8*lane-half
      atomicAdd(&sums[msub * 16 + lhalf * 8 + r], v);  // exactly 2 adds/slot -> deterministic
  }
  __syncthreads();
  if (tid < 64) scores[(size_t)(b0 + tid) * N_LBL + t] = sums[tid];
}

// =================== Kernel 2: WARP loss per batch row ===================
__global__ __launch_bounds__(128) void loss_rows_kernel(
    const float* __restrict__ scores, const int* __restrict__ target,
    float* __restrict__ row_loss)
{
  __shared__ float s[N_LBL];
  __shared__ int   negf[N_LBL];
  __shared__ float rwv[N_LBL];
  __shared__ float red[128];
  const int b = blockIdx.x, tid = threadIdx.x;
  if (tid < N_LBL) {
    s[tid] = scores[(size_t)b * N_LBL + tid];
    negf[tid] = (target[(size_t)b * N_LBL + tid] == 0) ? 1 : 0;
    double a = 1.0;                              // rw[0]=1; rw[i]=rw[i-1]+(1/i+1), fp64 like numpy
    for (int q = 1; q <= tid; ++q) a += 1.0 / (double)q + 1.0;
    rwv[tid] = (float)a;
  }
  __syncthreads();
  float contrib = 0.0f;
  if (tid < N_LBL) {
    const float sj = s[tid];
    int cnt = 0; float msum = 0.0f;
    for (int k = 0; k < N_LBL; ++k) {
      if (negf[k]) {
        float m = 1.0f + s[k] - sj;
        if (m > 0.0f) { cnt++; msum += m; }
      }
    }
    if (!negf[tid]) {                            // positive label (target is 0/1)
      int r = cnt - 1;
      if (r < 0) r += N_LBL;                     // Python (bigger-1) % n
      contrib = rwv[r] * msum;
    }
  }
  red[tid] = contrib;
  __syncthreads();
  for (int off = 64; off > 0; off >>= 1) {
    if (tid < off) red[tid] += red[tid + off];
    __syncthreads();
  }
  if (tid == 0) row_loss[b] = red[0];
}

// =================== Kernel 3: deterministic final reduction ===================
__global__ __launch_bounds__(256) void reduce_kernel(const float* __restrict__ row_loss,
                                                     float* __restrict__ out)
{
  __shared__ float red[256];
  const int tid = threadIdx.x;
  float a = 0.0f;
  for (int i = tid; i < BATCH; i += 256) a += row_loss[i];
  red[tid] = a;
  __syncthreads();
  for (int off = 128; off > 0; off >>= 1) {
    if (tid < off) red[tid] += red[tid + off];
    __syncthreads();
  }
  if (tid == 0) out[0] = red[0] * (1.0f / (float)BATCH);
}

// =================== host launch ===================
extern "C" void kernel_launch(void* const* d_in, const int* in_sizes, int n_in,
                              void* d_out, int out_size, void* d_ws, size_t ws_size,
                              hipStream_t stream) {
  const float* ctx  = (const float*)d_in[0];   // (89,1024,1024)
  const float* men  = (const float*)d_in[1];   // (1024,300)
  const float* te   = (const float*)d_in[2];   // (89,300)
  const float* W    = (const float*)d_in[3];   // (300,1324)
  const float* bv   = (const float*)d_in[4];   // (300,)
  const int*   tgt  = (const int*)d_in[5];     // (1024,89)

  char* ws = (char*)d_ws;
  const size_t SZ_WH  = (size_t)E_PAD * D_PAD * sizeof(unsigned short);   // 860160
  unsigned short* Whi = (unsigned short*)(ws);
  unsigned short* Wlo = (unsigned short*)(ws + SZ_WH);
  float* men_pad      = (float*)(ws + 2 * SZ_WH);                         // 1310720 B
  float* b_pad        = (float*)(ws + 2 * SZ_WH + (size_t)BATCH * MEN_PD * 4);
  float* te_pad       = (float*)((char*)b_pad + (size_t)E_PAD * 4);
  float* scores       = (float*)((char*)te_pad + (size_t)N_LBL * E_PAD * 4);
  float* row_loss     = (float*)((char*)scores + (size_t)BATCH * N_LBL * 4);

  const int NPREP = E_PAD * D_PAD + BATCH * MEN_PD + E_PAD + N_LBL * E_PAD;
  prep_kernel<<<(NPREP + 255) / 256, 256, 0, stream>>>(W, men, bv, te, Whi, Wlo,
                                                       men_pad, b_pad, te_pad);

  dim3 grid1(BATCH / 64, N_LBL);
  score_kernel<<<grid1, 256, 0, stream>>>(ctx, Whi, Wlo, men_pad, b_pad, te_pad, scores);

  loss_rows_kernel<<<BATCH, 128, 0, stream>>>(scores, tgt, row_loss);
  reduce_kernel<<<1, 256, 0, stream>>>(row_loss, (float*)d_out);
}
